// LSTMSequenceModel_90202903150772
// MI455X (gfx1250) — matrix operand order repaired and checked by reading, not compiled, not run on hardware
//
#include <hip/hip_runtime.h>
#include <hip/hip_bf16.h>

#define D_MODEL 1024
#define SEQ     1024
#define BATCH   16
#define GDIM    4096   // 4*D
#define MROWS   16384  // B*T

typedef __attribute__((ext_vector_type(16))) __bf16       v16bf;
typedef __attribute__((ext_vector_type(8)))  float        v8f;
typedef __attribute__((ext_vector_type(4)))  unsigned int uint4v;
typedef __attribute__((ext_vector_type(4)))  unsigned int u32x4;
typedef __attribute__((ext_vector_type(8)))  int          i32x8;
typedef __attribute__((ext_vector_type(4)))  int          i32x4;

union FragU { uint4v u[2]; v16bf v; };

__device__ __forceinline__ unsigned short f2bf(float f) {
  unsigned u = __float_as_uint(f);
  u += 0x7FFFu + ((u >> 16) & 1u);          // round-to-nearest-even
  return (unsigned short)(u >> 16);
}

// A-matrix fragment (16x32 bf16, ld=1024): lane l<16 -> row l, K {k0..k0+7, k0+16..k0+23}
__device__ __forceinline__ v16bf load_fragA(const unsigned short* __restrict__ base, int k0, int lane) {
  int l = lane & 15;
  int h = (lane >> 4) & 1;
  const unsigned short* p = base + l * D_MODEL + k0 + h * 8;
  FragU f;
  f.u[0] = *(const uint4v*)(p);
  f.u[1] = *(const uint4v*)(p + 16);
  return f.v;
}

// B-matrix fragment (32x16 bf16): col n of B = row n of row-major [N,1024] weight.
// lane l<16 -> col l, K {k0..k0+7, k0+16..k0+23}; lane l>=16 -> col l-16, K {k0+8..k0+15, k0+24..k0+31}.
__device__ __forceinline__ v16bf load_fragB(const unsigned short* __restrict__ base, int k0, int lane) {
  int l = lane & 15;
  int h = (lane >> 4) & 1;
  const unsigned short* p = base + l * D_MODEL + k0 + h * 8;
  FragU f;
  f.u[0] = *(const uint4v*)(p);
  f.u[1] = *(const uint4v*)(p + 16);
  return f.v;
}

// ---------------------------------------------------------------------------
// TDM: async 2D tile load global -> LDS (bf16 elements), per ISA ch.8 D# layout.
// Group0: count=1 | lds_addr | global_addr[56:0] | type=2.
// Group1: data_size=1(2B) | tensor_dim0/1 | tile_dim0/1 | tensor_dim0_stride.
// 6-arg builtin variant (clang-23 / therock headers).
// ---------------------------------------------------------------------------
__device__ __forceinline__ void tdm_load_tile_2d(unsigned lds_byte_off, const void* gaddr,
                                                 unsigned tile_d0, unsigned tile_d1,
                                                 unsigned tensor_d0, unsigned tensor_d1,
                                                 unsigned stride0) {
  unsigned long long ga = (unsigned long long)(__UINTPTR_TYPE__)gaddr;
  u32x4 g0;
  g0[0] = 1u;                                               // count=1, user descriptor
  g0[1] = lds_byte_off;                                     // lds_addr
  g0[2] = (unsigned)(ga & 0xFFFFFFFFu);                     // global_addr[31:0]
  g0[3] = (unsigned)((ga >> 32) & 0x01FFFFFFu) | (2u << 30);// global_addr[56:32] | type=2
  i32x8 g1;
  g1[0] = (int)(1u << 16);                                  // data_size = 2 bytes
  g1[1] = (int)(tensor_d0 << 16);                           // tensor_dim0[15:0]
  g1[2] = (int)((tensor_d0 >> 16) | (tensor_d1 << 16));     // tensor_dim0[31:16] | tensor_dim1[15:0]
  g1[3] = (int)((tensor_d1 >> 16) | (tile_d0 << 16));       // tensor_dim1[31:16] | tile_dim0
  g1[4] = (int)(tile_d1 & 0xFFFFu);                         // tile_dim1 (tile_dim2=0)
  g1[5] = (int)stride0;                                     // tensor_dim0_stride[31:0]
  g1[6] = 0;
  g1[7] = 0;
  i32x4 z4 = {0, 0, 0, 0};
  i32x8 z8 = {0, 0, 0, 0, 0, 0, 0, 0};
  __builtin_amdgcn_tensor_load_to_lds(g0, g1, z4, z4, z8, 0);
}

// ---------------------------------------------------------------------------
// fp32 -> bf16 weight conversion
// ---------------------------------------------------------------------------
__global__ void __launch_bounds__(256) f32_to_bf16_kernel(const float* __restrict__ in,
                                                          unsigned short* __restrict__ out, int n) {
  for (int i = blockIdx.x * 256 + threadIdx.x; i < n; i += gridDim.x * 256)
    out[i] = f2bf(in[i]);
}

// ---------------------------------------------------------------------------
// LayerNorm: one block per row, writes fp32 residual + bf16 copy for WMMA
// ---------------------------------------------------------------------------
__global__ void __launch_bounds__(256) ln_kernel(const float* __restrict__ x,
                                                 const float* __restrict__ gamma,
                                                 const float* __restrict__ beta,
                                                 float* __restrict__ res,
                                                 unsigned short* __restrict__ resb) {
  __shared__ float sred[8];
  const int row = blockIdx.x;
  const int tid = threadIdx.x;
  const float* xr = x + (size_t)row * D_MODEL;

  float v[4];
  float s = 0.f;
#pragma unroll
  for (int j = 0; j < 4; ++j) { v[j] = xr[tid + j * 256]; s += v[j]; }
#pragma unroll
  for (int o = 16; o > 0; o >>= 1) s += __shfl_down(s, o, 32);
  if ((tid & 31) == 0) sred[tid >> 5] = s;
  __syncthreads();
  if (tid == 0) { float t = 0.f; for (int i = 0; i < 8; ++i) t += sred[i]; sred[0] = t; }
  __syncthreads();
  const float mean = sred[0] * (1.f / 1024.f);
  __syncthreads();

  float s2 = 0.f;
#pragma unroll
  for (int j = 0; j < 4; ++j) { float d = v[j] - mean; s2 += d * d; }
#pragma unroll
  for (int o = 16; o > 0; o >>= 1) s2 += __shfl_down(s2, o, 32);
  if ((tid & 31) == 0) sred[tid >> 5] = s2;
  __syncthreads();
  if (tid == 0) { float t = 0.f; for (int i = 0; i < 8; ++i) t += sred[i]; sred[0] = t; }
  __syncthreads();
  const float rs = rsqrtf(sred[0] * (1.f / 1024.f) + 1e-5f);

#pragma unroll
  for (int j = 0; j < 4; ++j) {
    int c = tid + j * 256;
    float y = (v[j] - mean) * rs * gamma[c] + beta[c];
    size_t idx = (size_t)row * D_MODEL + c;
    res[idx]  = y;
    resb[idx] = f2bf(y);
  }
}

// ---------------------------------------------------------------------------
// Software-pipelined 16x64-per-wave GEMM body: acc += A[m0:16, :] * W[n0:64, :]^T
// Unroll-by-2 steady state so the double-buffer rotation becomes register
// renaming instead of v_mov copies.
// ---------------------------------------------------------------------------
__device__ __forceinline__ void gemm_body(const unsigned short* __restrict__ Abase,
                                          const unsigned short* __restrict__ W,
                                          int n0, int lane, v8f acc[4]) {
  v16bf a0 = load_fragA(Abase, 0, lane);
  v16bf b0[4];
#pragma unroll
  for (int j = 0; j < 4; ++j) b0[j] = load_fragB(W + (size_t)(n0 + j * 16) * D_MODEL, 0, lane);

#pragma unroll 2
  for (int kc = 0; kc < D_MODEL - 32; kc += 32) {
    v16bf a1 = load_fragA(Abase, kc + 32, lane);
    v16bf b1[4];
#pragma unroll
    for (int j = 0; j < 4; ++j) b1[j] = load_fragB(W + (size_t)(n0 + j * 16) * D_MODEL, kc + 32, lane);
#pragma unroll
    for (int j = 0; j < 4; ++j)
      acc[j] = __builtin_amdgcn_wmma_f32_16x16x32_bf16(false, a0, false, b0[j], (short)0, acc[j], false, false);
    a0 = a1;
#pragma unroll
    for (int j = 0; j < 4; ++j) b0[j] = b1[j];
  }
#pragma unroll
  for (int j = 0; j < 4; ++j)
    acc[j] = __builtin_amdgcn_wmma_f32_16x16x32_bf16(false, a0, false, b0[j], (short)0, acc[j], false, false);
}

// ---------------------------------------------------------------------------
// xp = res @ W_ih^T + (b_ih + b_hh).  M=16384, N=4096, K=1024.
// ---------------------------------------------------------------------------
__global__ void __launch_bounds__(256) gemm_xp_kernel(const unsigned short* __restrict__ A,
                                                      const unsigned short* __restrict__ W,
                                                      const float* __restrict__ bih,
                                                      const float* __restrict__ bhh,
                                                      float* __restrict__ xp) {
  const int lane = threadIdx.x & 31;
  const int gw = blockIdx.x * 8 + (threadIdx.x >> 5);
  const int m0 = (gw >> 6) * 16;
  const int n0 = (gw & 63) * 64;

  v8f acc[4] = {};
  gemm_body(A + (size_t)m0 * D_MODEL, W, n0, lane, acc);

  const int l = lane & 15;
  const int rowoff = (lane >> 4) * 8;
#pragma unroll
  for (int j = 0; j < 4; ++j) {
    int n = n0 + j * 16 + l;
    float bias = bih[n] + bhh[n];
#pragma unroll
    for (int r = 0; r < 8; ++r)
      xp[(size_t)(m0 + rowoff + r) * GDIM + n] = acc[j][r] + bias;
  }
}

// ---------------------------------------------------------------------------
// silu_lin = silu(res @ lin_W^T + lin_b).  M=16384, N=1024, K=1024.
// ---------------------------------------------------------------------------
__global__ void __launch_bounds__(256) gemm_lin_kernel(const unsigned short* __restrict__ A,
                                                       const unsigned short* __restrict__ W,
                                                       const float* __restrict__ lb,
                                                       float* __restrict__ out) {
  const int lane = threadIdx.x & 31;
  const int gw = blockIdx.x * 8 + (threadIdx.x >> 5);
  const int m0 = (gw >> 4) * 16;
  const int n0 = (gw & 15) * 64;

  v8f acc[4] = {};
  gemm_body(A + (size_t)m0 * D_MODEL, W, n0, lane, acc);

  const int l = lane & 15;
  const int rowoff = (lane >> 4) * 8;
#pragma unroll
  for (int j = 0; j < 4; ++j) {
    int n = n0 + j * 16 + l;
    float bias = lb[n];
#pragma unroll
    for (int r = 0; r < 8; ++r) {
      float s = acc[j][r] + bias;
      out[(size_t)(m0 + rowoff + r) * D_MODEL + n] = s / (1.f + __expf(-s));   // silu
    }
  }
}

// ---------------------------------------------------------------------------
// Persistent LSTM recurrence. 32 blocks x 64 threads (2 waves). Wave w owns
// d-columns [16w,16w+16); its 4 gate weight tiles (16 rows x 1024 K each, bf16)
// are DMA'd into LDS by the Tensor Data Mover (128 KB/wave, 256 KB/block).
// h double-buffered in global; grid-wide release/acquire barrier per step.
// Epilogue fuses out = h*silu(lin) + res.
// ---------------------------------------------------------------------------
__global__ void __launch_bounds__(64) lstm_seq_kernel(const float* __restrict__ xp,
                                                      const unsigned short* __restrict__ Whh,
                                                      const float* __restrict__ gate_silu,
                                                      const float* __restrict__ res,
                                                      unsigned short* __restrict__ h0,
                                                      unsigned short* __restrict__ h1,
                                                      float* __restrict__ out,
                                                      unsigned* __restrict__ counter) {
  extern __shared__ unsigned short smem[];
  const int lane = threadIdx.x & 31;
  const int wv = __builtin_amdgcn_readfirstlane(threadIdx.x >> 5);
  const int w = blockIdx.x * 2 + wv;      // 0..63 d-tile
  const int d0 = w * 16;
  unsigned short* sW = smem + (size_t)wv * (64 * D_MODEL);

  // Stage W_hh rows for i,f,g,o into LDS via TDM (one 16x1024 bf16 tile per gate).
#pragma unroll
  for (int g = 0; g < 4; ++g) {
    unsigned ldsoff = (unsigned)(wv * (64 * D_MODEL) + g * 16 * D_MODEL) * 2u;
    tdm_load_tile_2d(ldsoff,
                     Whh + (size_t)(g * D_MODEL + d0) * D_MODEL,
                     /*tile_d0=*/D_MODEL, /*tile_d1=*/16,
                     /*tensor_d0=*/D_MODEL, /*tensor_d1=*/GDIM,
                     /*stride0=*/D_MODEL);
  }
  __builtin_amdgcn_s_wait_tensorcnt(0);
  __syncthreads();

  v8f cstate = {};
  const int l = lane & 15;
  const int rowoff = (lane >> 4) * 8;
  const unsigned nblocks = gridDim.x;

  for (int t = 0; t < SEQ; ++t) {
    const unsigned short* hin = (t & 1) ? h1 : h0;
    unsigned short* hout = (t & 1) ? h0 : h1;

    v8f acc[4] = {};
    v16bf a0 = load_fragA(hin, 0, lane);                   // h: [16,1024] bf16
#pragma unroll 2
    for (int kc = 0; kc < D_MODEL - 32; kc += 32) {
      v16bf a1 = load_fragA(hin, kc + 32, lane);
#pragma unroll
      for (int g = 0; g < 4; ++g)
        acc[g] = __builtin_amdgcn_wmma_f32_16x16x32_bf16(
            false, a0, false, load_fragB(sW + g * 16 * D_MODEL, kc, lane), (short)0, acc[g], false, false);
      a0 = a1;
    }
#pragma unroll
    for (int g = 0; g < 4; ++g)
      acc[g] = __builtin_amdgcn_wmma_f32_16x16x32_bf16(
          false, a0, false, load_fragB(sW + g * 16 * D_MODEL, D_MODEL - 32, lane), (short)0, acc[g], false, false);

    const int d = d0 + l;
#pragma unroll
    for (int r = 0; r < 8; ++r) {
      int b = rowoff + r;                                  // batch index
      size_t gbase = ((size_t)b * SEQ + t) * GDIM;
      float gi = acc[0][r] + xp[gbase + 0 * D_MODEL + d];
      float gf = acc[1][r] + xp[gbase + 1 * D_MODEL + d];
      float gg = acc[2][r] + xp[gbase + 2 * D_MODEL + d];
      float go = acc[3][r] + xp[gbase + 3 * D_MODEL + d];
      float iv = 1.f / (1.f + __expf(-gi));
      float fv = 1.f / (1.f + __expf(-gf));
      float gv = tanhf(gg);
      float ov = 1.f / (1.f + __expf(-go));
      float cv = fv * cstate[r] + iv * gv;
      cstate[r] = cv;
      float hv = ov * tanhf(cv);
      hout[b * D_MODEL + d] = f2bf(hv);
      size_t oidx = ((size_t)b * SEQ + t) * D_MODEL + d;
      out[oidx] = hv * gate_silu[oidx] + res[oidx];
    }

    // grid-wide barrier (release/acquire, device scope)
    __syncthreads();
    if (threadIdx.x == 0) {
      __hip_atomic_fetch_add(counter, 1u, __ATOMIC_RELEASE, __HIP_MEMORY_SCOPE_AGENT);
      unsigned target = nblocks * (unsigned)(t + 1);
      while (__hip_atomic_load(counter, __ATOMIC_ACQUIRE, __HIP_MEMORY_SCOPE_AGENT) < target)
        __builtin_amdgcn_s_sleep(1);
    }
    __syncthreads();
  }
}

// ---------------------------------------------------------------------------
extern "C" void kernel_launch(void* const* d_in, const int* in_sizes, int n_in,
                              void* d_out, int out_size, void* d_ws, size_t ws_size,
                              hipStream_t stream) {
  const float* x     = (const float*)d_in[0];
  const float* W_ih  = (const float*)d_in[1];
  const float* W_hh  = (const float*)d_in[2];
  const float* b_ih  = (const float*)d_in[3];
  const float* b_hh  = (const float*)d_in[4];
  const float* ln_g  = (const float*)d_in[5];
  const float* ln_b  = (const float*)d_in[6];
  const float* lin_W = (const float*)d_in[7];
  const float* lin_b = (const float*)d_in[8];
  float* outp = (float*)d_out;

  char* ws = (char*)d_ws;
  size_t off = 0;
  auto alloc = [&](size_t bytes) -> void* {
    void* p = ws + off;
    off = (off + bytes + 255) & ~(size_t)255;
    return p;
  };

  unsigned* counter     = (unsigned*)alloc(256);
  unsigned short* hbuf  = (unsigned short*)alloc((size_t)2 * BATCH * D_MODEL * 2);
  float* resf           = (float*)alloc((size_t)MROWS * D_MODEL * 4);
  unsigned short* resb  = (unsigned short*)alloc((size_t)MROWS * D_MODEL * 2);
  float* xp             = (float*)alloc((size_t)MROWS * GDIM * 4);
  float* silu           = (float*)alloc((size_t)MROWS * D_MODEL * 4);
  float* xbuf           = (float*)alloc((size_t)MROWS * D_MODEL * 4);
  unsigned short* wihb  = (unsigned short*)alloc((size_t)GDIM * D_MODEL * 2);
  unsigned short* whhb  = (unsigned short*)alloc((size_t)GDIM * D_MODEL * 2);
  unsigned short* linwb = (unsigned short*)alloc((size_t)D_MODEL * D_MODEL * 2);

  for (int ll = 0; ll < 4; ++ll) {
    const float* xin = (ll == 0) ? x : xbuf;
    float* xout = (ll == 3) ? outp : xbuf;

    ln_kernel<<<MROWS, 256, 0, stream>>>(xin, ln_g + ll * D_MODEL, ln_b + ll * D_MODEL, resf, resb);

    f32_to_bf16_kernel<<<4096, 256, 0, stream>>>(W_ih  + (size_t)ll * GDIM * D_MODEL, wihb,  GDIM * D_MODEL);
    f32_to_bf16_kernel<<<4096, 256, 0, stream>>>(W_hh  + (size_t)ll * GDIM * D_MODEL, whhb,  GDIM * D_MODEL);
    f32_to_bf16_kernel<<<1024, 256, 0, stream>>>(lin_W + (size_t)ll * D_MODEL * D_MODEL, linwb, D_MODEL * D_MODEL);

    gemm_xp_kernel<<<8192, 256, 0, stream>>>(resb, wihb, b_ih + ll * GDIM, b_hh + ll * GDIM, xp);
    gemm_lin_kernel<<<2048, 256, 0, stream>>>(resb, linwb, lin_b + ll * D_MODEL, silu);

    // zero barrier counter + both h buffers (contiguous at ws base)
    (void)hipMemsetAsync(counter, 0, 256 + (size_t)2 * BATCH * D_MODEL * 2, stream);

    lstm_seq_kernel<<<32, 64, 2 * 64 * D_MODEL * sizeof(unsigned short), stream>>>(
        xp, whhb, silu, resf, hbuf, hbuf + BATCH * D_MODEL, xout, counter);
  }
}
